// FlowODE_75797582839847
// MI455X (gfx1250) — matrix operand order repaired and checked by reading, not compiled, not run on hardware
//
#include <hip/hip_runtime.h>
#include <hip/hip_bf16.h>
#include <math.h>
#include <stdint.h>

typedef __attribute__((ext_vector_type(16))) __bf16 v16bf;
typedef __attribute__((ext_vector_type(8)))  float  v8f;
typedef int v4i_async __attribute__((vector_size(16)));

#define MTOK 4096
#define DDIM 2048
#define RDIM 256
#define HDIM 1024
#define VDIM 32000
#define NSTEPS 32
#define RMS_EPSF 1.1920929e-07f

// ---------------------------------------------------------------------------
// CDNA5 async global->LDS copy (ASYNCcnt-tracked), with sync fallback.
// Signature per hipcc diagnostic: (v4i* src, v4i* dst, imm offset, imm cpol),
// generic pointers; wait counter must be a literal constant.
// ---------------------------------------------------------------------------
#if defined(__has_builtin)
#if __has_builtin(__builtin_amdgcn_global_load_async_to_lds_b128) && \
    __has_builtin(__builtin_amdgcn_s_wait_asynccnt)
#define USE_ASYNC_LDS 1
#endif
#endif
#ifndef USE_ASYNC_LDS
#define USE_ASYNC_LDS 0
#endif

__device__ __forceinline__ void copy16_to_lds(const void* g, void* l) {
#if USE_ASYNC_LDS
  __builtin_amdgcn_global_load_async_to_lds_b128(
      (v4i_async*)g, (v4i_async*)l, 0, 0);
#else
  *(uint4*)l = *(const uint4*)g;
#endif
}

__device__ __forceinline__ void wait_async_lds_keep1() {
#if USE_ASYNC_LDS
  __builtin_amdgcn_s_wait_asynccnt(1);
#endif
}

__device__ __forceinline__ void wait_async_lds_all() {
#if USE_ASYNC_LDS
  __builtin_amdgcn_s_wait_asynccnt(0);
#endif
}

// ---------------------------------------------------------------------------
// f32 -> bf16 conversion (grid-stride)
// ---------------------------------------------------------------------------
__global__ void cvt_bf16_kernel(const float* __restrict__ in,
                                __bf16* __restrict__ out, int n) {
  int i = blockIdx.x * blockDim.x + threadIdx.x;
  int st = gridDim.x * blockDim.x;
  for (; i < n; i += st) out[i] = (__bf16)in[i];
}

// P [R, D] f32 -> PT [D, R] bf16 (so lift-GEMM B fragments are contiguous)
__global__ void transpose_P_kernel(const float* __restrict__ P,
                                   __bf16* __restrict__ PT) {
  int i = blockIdx.x * blockDim.x + threadIdx.x;
  int st = gridDim.x * blockDim.x;
  for (; i < RDIM * DDIM; i += st) {
    int r = i / DDIM, d = i % DDIM;
    PT[(size_t)d * RDIM + r] = (__bf16)P[i];
  }
}

// ---------------------------------------------------------------------------
// Fused embedding gather + projection:  z = x @ P^T  (M=4096, N=256, K=2048)
// grid (32, 2) x 256 threads; each wave owns a 16-row M-tile and an N-half
// (8 tiles -> 64 accumulator VGPRs, no spills).  y==0 blocks also store x
// via vectorized b128 stores sharing the b128 loads with the WMMA A path.
// ---------------------------------------------------------------------------
__global__ __launch_bounds__(256) void embed_project_kernel(
    const int* __restrict__ tokens, const float* __restrict__ embed_w,
    const __bf16* __restrict__ Pb, float* __restrict__ x_out,
    float* __restrict__ z_out) {
  const int wave = threadIdx.x >> 5;
  const int lane = threadIdx.x & 31;
  const int mrow = lane & 15;   // M row of this lane (A/B operand row/col)
  const int kh   = lane >> 4;   // lane half selects K sub-ranges
  const int m0   = blockIdx.x * 128 + wave * 16;
  const int ntb  = blockIdx.y * 8;          // n-tile base
  const bool writex = (blockIdx.y == 0);

  const int tok = tokens[m0 + mrow];
  const float* src  = embed_w + (size_t)tok * DDIM;
  float*       xdst = x_out + (size_t)(m0 + mrow) * DDIM;

  v8f acc[8];
#pragma unroll
  for (int nt = 0; nt < 8; ++nt)
#pragma unroll
    for (int j = 0; j < 8; ++j) acc[nt][j] = 0.0f;

  for (int k0 = 0; k0 < DDIM; k0 += 32) {
    // A fragment (16x32 bf16): half-lane kh=0 holds K {0..7,16..23},
    // kh=1 holds K {8..15,24..31}; load as four b128 packets.
    const float4 q0 = *(const float4*)(src + k0 + kh * 8);
    const float4 q1 = *(const float4*)(src + k0 + kh * 8 + 4);
    const float4 q2 = *(const float4*)(src + k0 + 16 + kh * 8);
    const float4 q3 = *(const float4*)(src + k0 + 16 + kh * 8 + 4);
    if (writex) {  // store x exactly once per element, b128 stores
      *(float4*)(xdst + k0 + kh * 8)          = q0;
      *(float4*)(xdst + k0 + kh * 8 + 4)      = q1;
      *(float4*)(xdst + k0 + 16 + kh * 8)     = q2;
      *(float4*)(xdst + k0 + 16 + kh * 8 + 4) = q3;
    }
    v16bf af;
    af[0]  = (__bf16)q0.x; af[1]  = (__bf16)q0.y;
    af[2]  = (__bf16)q0.z; af[3]  = (__bf16)q0.w;
    af[4]  = (__bf16)q1.x; af[5]  = (__bf16)q1.y;
    af[6]  = (__bf16)q1.z; af[7]  = (__bf16)q1.w;
    af[8]  = (__bf16)q2.x; af[9]  = (__bf16)q2.y;
    af[10] = (__bf16)q2.z; af[11] = (__bf16)q2.w;
    af[12] = (__bf16)q3.x; af[13] = (__bf16)q3.y;
    af[14] = (__bf16)q3.z; af[15] = (__bf16)q3.w;

#pragma unroll
    for (int nt = 0; nt < 8; ++nt) {
      // B[k][n] = P[n][k]; lane n reads K {kh*8..+7, 16+kh*8..+7} of row n
      const __bf16* bp =
          Pb + (size_t)((ntb + nt) * 16 + mrow) * DDIM + k0 + kh * 8;
      v16bf bf;
#pragma unroll
      for (int i = 0; i < 16; ++i) bf[i] = bp[(i & 7) + (i >> 3) * 16];
      acc[nt] = __builtin_amdgcn_wmma_f32_16x16x32_bf16(
          false, af, false, bf, (short)0, acc[nt], false, false);
    }
  }
  // C layout: vgpr j, lane -> (m = j + 8*kh, n = mrow)
#pragma unroll
  for (int nt = 0; nt < 8; ++nt)
#pragma unroll
    for (int j = 0; j < 8; ++j)
      z_out[(size_t)(m0 + j + 8 * kh) * RDIM + (ntb + nt) * 16 + mrow] =
          acc[nt][j];
}

// ---------------------------------------------------------------------------
// ODE: all 32 steps in one kernel.  Rows are independent; each wave keeps its
// 16x256 z tile in registers (C layout, f32 master), republishes bf16 A
// fragments via LDS each step, streams h in 32-column chunks through LDS with
// exact-erf GELU.  Weights w1/w2 are ~1MB bf16 total -> L2 resident.
// grid 64 blocks x 128 threads (4 waves, 1 wave/SIMD for register headroom).
// ---------------------------------------------------------------------------
__global__ __launch_bounds__(128) void ode_kernel(
    float* __restrict__ z, const __bf16* __restrict__ w1b,
    const float* __restrict__ b1, const __bf16* __restrict__ w2b,
    const float* __restrict__ b2, const float* __restrict__ log_tau) {
  extern __shared__ char smem[];
  __bf16* zlds = (__bf16*)smem;                           // 4*16*256 bf16
  __bf16* hlds = (__bf16*)(smem + 4 * 16 * RDIM * 2);     // 4*16*32  bf16

  const int wave = threadIdx.x >> 5;
  const int lane = threadIdx.x & 31;
  const int mrow = lane & 15;
  const int kh   = lane >> 4;
  const int m0   = blockIdx.x * 64 + wave * 16;
  __bf16* zw = zlds + wave * 16 * RDIM;
  __bf16* hw = hlds + wave * 16 * 32;

  v8f zc[16];  // z master, C layout
#pragma unroll
  for (int nt = 0; nt < 16; ++nt)
#pragma unroll
    for (int j = 0; j < 8; ++j)
      zc[nt][j] = z[(size_t)(m0 + j + 8 * kh) * RDIM + nt * 16 + mrow];

#pragma unroll 1
  for (int s = 0; s < NSTEPS; ++s) {
    const float t01 = __expf(log_tau[s]) * 0.01f;

    // publish z to LDS as bf16 (per-wave region; wave-ordered DS ops)
#pragma unroll
    for (int nt = 0; nt < 16; ++nt)
#pragma unroll
      for (int j = 0; j < 8; ++j)
        zw[(j + 8 * kh) * RDIM + nt * 16 + mrow] = (__bf16)zc[nt][j];

    // hoist all 8 z A-fragments for this step
    v16bf afz[8];
#pragma unroll
    for (int kr = 0; kr < 8; ++kr) {
      const __bf16* p0 = zw + mrow * RDIM + kr * 32 + kh * 8;
      const __bf16* p1 = zw + mrow * RDIM + kr * 32 + 16 + kh * 8;
#pragma unroll
      for (int i = 0; i < 8; ++i) { afz[kr][i] = p0[i]; afz[kr][8 + i] = p1[i]; }
    }

    v8f dz[16];
#pragma unroll
    for (int nt = 0; nt < 16; ++nt)
#pragma unroll
      for (int j = 0; j < 8; ++j) dz[nt][j] = 0.0f;

#pragma unroll 1
    for (int c = 0; c < 32; ++c) {  // 32-column chunk of h (H = 1024)
      v8f hacc[2];
#pragma unroll
      for (int t = 0; t < 2; ++t) {
        float bv = b1[c * 32 + t * 16 + mrow];
#pragma unroll
        for (int j = 0; j < 8; ++j) hacc[t][j] = bv;
      }
#pragma unroll
      for (int kr = 0; kr < 8; ++kr)
#pragma unroll
        for (int t = 0; t < 2; ++t) {
          // B[k][n] = w1[h=n][k]; K {kh*8..+7, 16+kh*8..+7} per lane
          const __bf16* bp =
              w1b + (size_t)(c * 32 + t * 16 + mrow) * RDIM + kr * 32 + kh * 8;
          v16bf bf;
#pragma unroll
          for (int i = 0; i < 16; ++i) bf[i] = bp[(i & 7) + (i >> 3) * 16];
          hacc[t] = __builtin_amdgcn_wmma_f32_16x16x32_bf16(
              false, afz[kr], false, bf, (short)0, hacc[t], false, false);
        }
      // exact GELU, stage chunk to LDS (C layout -> row-major bf16)
#pragma unroll
      for (int t = 0; t < 2; ++t)
#pragma unroll
        for (int j = 0; j < 8; ++j) {
          float xv = hacc[t][j];
          float g  = 0.5f * xv * (1.0f + erff(xv * 0.70710678118f));
          hw[(j + 8 * kh) * 32 + t * 16 + mrow] = (__bf16)g;
        }
      // h chunk as A operand (K = 32 slice of H)
      v16bf afh;
      {
        const __bf16* p0 = hw + mrow * 32 + kh * 8;
        const __bf16* p1 = hw + mrow * 32 + 16 + kh * 8;
#pragma unroll
        for (int i = 0; i < 8; ++i) { afh[i] = p0[i]; afh[8 + i] = p1[i]; }
      }
#pragma unroll
      for (int nt = 0; nt < 16; ++nt) {
        // B[k][n] = w2[r=n][h=k]; K {kh*8..+7, 16+kh*8..+7} per lane
        const __bf16* bp =
            w2b + (size_t)(nt * 16 + mrow) * HDIM + c * 32 + kh * 8;
        v16bf bf;
#pragma unroll
        for (int i = 0; i < 16; ++i) bf[i] = bp[(i & 7) + (i >> 3) * 16];
        dz[nt] = __builtin_amdgcn_wmma_f32_16x16x32_bf16(
            false, afh, false, bf, (short)0, dz[nt], false, false);
      }
    }
    // z += tau*0.01*(dz + b2)
#pragma unroll
    for (int nt = 0; nt < 16; ++nt) {
      float b2v = b2[nt * 16 + mrow];
#pragma unroll
      for (int j = 0; j < 8; ++j) zc[nt][j] += t01 * (dz[nt][j] + b2v);
    }
  }
#pragma unroll
  for (int nt = 0; nt < 16; ++nt)
#pragma unroll
    for (int j = 0; j < 8; ++j)
      z[(size_t)(m0 + j + 8 * kh) * RDIM + nt * 16 + mrow] = zc[nt][j];
}

// ---------------------------------------------------------------------------
// Lift: x_out = x + z @ P  (M=4096, N=2048, K=256) + per-row sum of squares.
// grid 32 x 256.
// ---------------------------------------------------------------------------
__global__ __launch_bounds__(256) void lift_rms_kernel(
    const float* __restrict__ z, const __bf16* __restrict__ PTb,
    const float* __restrict__ xbuf, float* __restrict__ xout,
    float* __restrict__ rss) {
  const int wave = threadIdx.x >> 5;
  const int lane = threadIdx.x & 31;
  const int mrow = lane & 15, kh = lane >> 4;
  const int m0 = blockIdx.x * 128 + wave * 16;

  v16bf afz[8];  // z A-fragments (K = 256), converted f32->bf16 on the fly
#pragma unroll
  for (int kr = 0; kr < 8; ++kr) {
    const float* p0 = z + (size_t)(m0 + mrow) * RDIM + kr * 32 + kh * 8;
    const float* p1 = z + (size_t)(m0 + mrow) * RDIM + kr * 32 + 16 + kh * 8;
#pragma unroll
    for (int i = 0; i < 8; ++i) {
      afz[kr][i]     = (__bf16)p0[i];
      afz[kr][8 + i] = (__bf16)p1[i];
    }
  }
  float psum[8];
#pragma unroll
  for (int j = 0; j < 8; ++j) psum[j] = 0.0f;

  for (int d0 = 0; d0 < DDIM; d0 += 64) {
    v8f acc[4];
#pragma unroll
    for (int nt = 0; nt < 4; ++nt)
#pragma unroll
      for (int j = 0; j < 8; ++j) acc[nt][j] = 0.0f;
#pragma unroll
    for (int kr = 0; kr < 8; ++kr)
#pragma unroll
      for (int nt = 0; nt < 4; ++nt) {
        // B[k][n] = P[k][d0+n] = PT[d0+n][k]; K {kh*8..+7, 16+kh*8..+7} per lane
        const __bf16* bp =
            PTb + (size_t)(d0 + nt * 16 + mrow) * RDIM + kr * 32 + kh * 8;
        v16bf bf;
#pragma unroll
        for (int i = 0; i < 16; ++i) bf[i] = bp[(i & 7) + (i >> 3) * 16];
        acc[nt] = __builtin_amdgcn_wmma_f32_16x16x32_bf16(
            false, afz[kr], false, bf, (short)0, acc[nt], false, false);
      }
#pragma unroll
    for (int nt = 0; nt < 4; ++nt)
#pragma unroll
      for (int j = 0; j < 8; ++j) {
        size_t idx = (size_t)(m0 + j + 8 * kh) * DDIM + d0 + nt * 16 + mrow;
        float v = xbuf[idx] + acc[nt][j];
        xout[idx] = v;
        psum[j] += v * v;
      }
  }
  // reduce across the 16 lanes sharing each row, write per-row sumsq
#pragma unroll
  for (int j = 0; j < 8; ++j) {
    float v = psum[j];
    for (int off = 8; off > 0; off >>= 1) v += __shfl_xor(v, off, 16);
    if (mrow == 0) rss[m0 + j + 8 * kh] = v;
  }
}

// RMSNorm scale + bf16 convert for the LM head A operand
__global__ void rmsnorm_kernel(const float* __restrict__ xout,
                               const float* __restrict__ rss,
                               const float* __restrict__ norm_w,
                               __bf16* __restrict__ xn) {
  int i = blockIdx.x * blockDim.x + threadIdx.x;
  int st = gridDim.x * blockDim.x;
  for (; i < MTOK * DDIM; i += st) {
    int m = i / DDIM, d = i % DDIM;
    float r = rsqrtf(rss[m] * (1.0f / DDIM) + RMS_EPSF);
    xn[i] = (__bf16)(xout[i] * r * norm_w[d]);
  }
}

// ---------------------------------------------------------------------------
// LM head: logits = x_n @ lm_head^T  (M=4096, N=32000, K=2048), bf16 weights
// L2-resident (131 MB < 192 MB).  grid (32 M-blocks fast, 500 N-blocks slow)
// so concurrent blocks share one weight stripe in L2.  The block-shared
// 64x32 weight chunk is staged into LDS via async global->LDS copies
// (double-buffered, ASYNCcnt + barrier synchronized); all 8 waves consume it
// from LDS instead of redundantly loading it 8x from global.
// ---------------------------------------------------------------------------
__global__ __launch_bounds__(256) void lmhead_kernel(
    const __bf16* __restrict__ xn, const __bf16* __restrict__ lmb,
    float* __restrict__ out) {
  __shared__ __bf16 bstage[2][64 * 32];  // 2 x 4 KB double buffer
  const int tid  = threadIdx.x;
  const int wave = tid >> 5;
  const int lane = tid & 31;
  const int mrow = lane & 15, kh = lane >> 4;
  const int m0 = blockIdx.x * 128 + wave * 16;
  const int n0 = blockIdx.y * 64;

  // copy role: each of 256 threads moves one 16B packet per k-chunk
  const int crow = tid >> 2;        // 0..63  (N row within chunk)
  const int ccol = (tid & 3) * 8;   // bf16 column offset within 32

  v8f acc[4];
#pragma unroll
  for (int nt = 0; nt < 4; ++nt)
#pragma unroll
    for (int j = 0; j < 8; ++j) acc[nt][j] = 0.0f;

  // prologue: stage chunk 0 into buffer 0
  copy16_to_lds(lmb + (size_t)(n0 + crow) * DDIM + ccol,
                &bstage[0][crow * 32 + ccol]);

#pragma unroll 1
  for (int kc = 0; kc < 64; ++kc) {
    const int k0 = kc * 32;
    if (kc + 1 < 64)
      copy16_to_lds(lmb + (size_t)(n0 + crow) * DDIM + (k0 + 32) + ccol,
                    &bstage[(kc + 1) & 1][crow * 32 + ccol]);
    if (kc < 63)
      wait_async_lds_keep1();  // chunk kc landed; next chunk stays in flight
    else
      wait_async_lds_all();    // final chunk: drain everything
    __syncthreads();           // ... and everyone else's copies

    const __bf16* bb = &bstage[kc & 1][0];

    v16bf af;
    const __bf16* p0 = xn + (size_t)(m0 + mrow) * DDIM + k0 + kh * 8;
    const __bf16* p1 = p0 + 16;
#pragma unroll
    for (int i = 0; i < 8; ++i) { af[i] = p0[i]; af[8 + i] = p1[i]; }
#pragma unroll
    for (int nt = 0; nt < 4; ++nt) {
      const __bf16* bp = bb + (nt * 16 + mrow) * 32 + kh * 8;
      v16bf bf;
#pragma unroll
      for (int i = 0; i < 16; ++i) bf[i] = bp[(i & 7) + (i >> 3) * 16];
      acc[nt] = __builtin_amdgcn_wmma_f32_16x16x32_bf16(
          false, af, false, bf, (short)0, acc[nt], false, false);
    }
    __syncthreads();  // buffer may be overwritten next iteration
  }
#pragma unroll
  for (int nt = 0; nt < 4; ++nt)
#pragma unroll
    for (int j = 0; j < 8; ++j)
      out[(size_t)(m0 + j + 8 * kh) * VDIM + n0 + nt * 16 + mrow] =
          acc[nt][j];
}

// ---------------------------------------------------------------------------
extern "C" void kernel_launch(void* const* d_in, const int* in_sizes, int n_in,
                              void* d_out, int out_size, void* d_ws,
                              size_t ws_size, hipStream_t stream) {
  (void)in_sizes; (void)n_in; (void)out_size; (void)ws_size;
  const int*   tokens  = (const int*)d_in[0];
  const float* P       = (const float*)d_in[1];
  /* d_in[2] = Q == P^T, unused (we use P directly) */
  const float* w1      = (const float*)d_in[3];
  const float* b1      = (const float*)d_in[4];
  const float* w2      = (const float*)d_in[5];
  const float* b2      = (const float*)d_in[6];
  const float* log_tau = (const float*)d_in[7];
  const float* embed_w = (const float*)d_in[8];
  const float* lm_head = (const float*)d_in[9];
  const float* norm_w  = (const float*)d_in[10];
  float* out = (float*)d_out;

  char* ws = (char*)d_ws;
  size_t off = 0;
  auto alloc = [&](size_t bytes) -> void* {
    off = (off + 255) & ~(size_t)255;
    void* p = ws + off;
    off += bytes;
    return p;
  };
  float*  ws_x    = (float*) alloc((size_t)MTOK * DDIM * 4);
  float*  ws_z    = (float*) alloc((size_t)MTOK * RDIM * 4);
  float*  ws_xout = (float*) alloc((size_t)MTOK * DDIM * 4);
  float*  ws_rss  = (float*) alloc((size_t)MTOK * 4);
  __bf16* ws_xn   = (__bf16*)alloc((size_t)MTOK * DDIM * 2);
  __bf16* ws_lmb  = (__bf16*)alloc((size_t)VDIM * DDIM * 2);
  __bf16* ws_Pb   = (__bf16*)alloc((size_t)RDIM * DDIM * 2);
  __bf16* ws_PTb  = (__bf16*)alloc((size_t)DDIM * RDIM * 2);
  __bf16* ws_w1b  = (__bf16*)alloc((size_t)HDIM * RDIM * 2);
  __bf16* ws_w2b  = (__bf16*)alloc((size_t)RDIM * HDIM * 2);

  cvt_bf16_kernel<<<2048, 256, 0, stream>>>(lm_head, ws_lmb, VDIM * DDIM);
  cvt_bf16_kernel<<<256, 256, 0, stream>>>(P, ws_Pb, RDIM * DDIM);
  cvt_bf16_kernel<<<128, 256, 0, stream>>>(w1, ws_w1b, HDIM * RDIM);
  cvt_bf16_kernel<<<128, 256, 0, stream>>>(w2, ws_w2b, RDIM * HDIM);
  transpose_P_kernel<<<256, 256, 0, stream>>>(P, ws_PTb);

  dim3 gproj(32, 2);
  embed_project_kernel<<<gproj, 256, 0, stream>>>(tokens, embed_w, ws_Pb, ws_x,
                                                  ws_z);
  const int ode_lds = 4 * 16 * RDIM * 2 + 4 * 16 * 32 * 2;  // 36 KB
  ode_kernel<<<64, 128, ode_lds, stream>>>(ws_z, ws_w1b, b1, ws_w2b, b2,
                                           log_tau);
  lift_rms_kernel<<<32, 256, 0, stream>>>(ws_z, ws_PTb, ws_x, ws_xout, ws_rss);
  rmsnorm_kernel<<<2048, 256, 0, stream>>>(ws_xout, ws_rss, norm_w, ws_xn);
  dim3 g6(32, 500);
  lmhead_kernel<<<g6, 256, 0, stream>>>(ws_xn, ws_lmb, out);
}
